// HybridQuantumBinaryClassifier_65481071402138
// MI455X (gfx1250) — matrix operand order repaired and checked by reading, not compiled, not run on hardware
//
#include <hip/hip_runtime.h>
#include <hip/hip_bf16.h>

// ---------------------------------------------------------------------------
// HybridQuantumBinaryClassifier forward on MI455X (gfx1250, wave32, WMMA)
//   conv1+relu (fp32 math, bf16 out) -> pool1 -> conv2+relu -> pool2(+pad,bf16)
//   fc1: bf16 WMMA GEMM (v_wmma_f32_16x16x32_bf16), 8-way K-split per block,
//        deterministic LDS reduction, fp32 accum, +bias, relu
//   fc2 (scalar fp32) -> fc3 + 4-qubit statevector head + softmax
// ---------------------------------------------------------------------------

typedef unsigned int u32x4 __attribute__((ext_vector_type(4)));
typedef __bf16       v16bf __attribute__((ext_vector_type(16)));
typedef float        v8f   __attribute__((ext_vector_type(8)));
typedef __hip_bfloat16 bf16_t;

// ---- shapes ----------------------------------------------------------------
constexpr int BATCH = 256;
constexpr int C1_OC = 6,  C1_H = 124, C1_W = 124;   // conv1 out
constexpr int P1_H  = 123, P1_W = 123;              // pool1 out
constexpr int C2_OC = 15, C2_H = 62,  C2_W = 62;    // conv2 out
constexpr int P2_H  = 61,  P2_W = 61;               // pool2 out
constexpr int K_FLAT = C2_OC * P2_H * P2_W;         // 55815
constexpr int KP     = 55872;                       // padded K, multiple of 64
constexpr int FC1_N  = 120, FC2_N = 84;
constexpr int KW     = 8;                           // waves per fc1 block (K-split)

constexpr int N_C1 = BATCH * C1_OC * C1_H * C1_W;   // 23,617,536
constexpr int N_P1 = BATCH * C1_OC * P1_H * P1_W;   // 23,238,144
constexpr int N_C2 = BATCH * C2_OC * C2_H * C2_W;   // 14,760,960
constexpr int N_HB = BATCH * KP;                    // 14,303,232
constexpr int N_WB = 128 * KP;                      // 7,151,616 (N padded 120->128)

// ---- workspace layout (two ping-pong regions, peak ~94 MB) ----------------
constexpr size_t AL(size_t x) { return (x + 255) & ~size_t(255); }
constexpr size_t OFF_C1  = 0;                                  // conv1 out / conv2 out
constexpr size_t OFF_P1  = AL((size_t)N_C1 * 2);               // pool1 out / fc1 acts
constexpr size_t OFF_WB  = OFF_P1 + AL((size_t)N_HB * 2);      // fc1 weights bf16
constexpr size_t OFF_FC1 = OFF_P1 + AL((size_t)N_P1 * 2);      // fc1 out fp32
constexpr size_t OFF_FC2 = OFF_FC1 + AL((size_t)BATCH * FC1_N * 4);
constexpr size_t WS_NEED = OFF_FC2 + AL((size_t)BATCH * FC2_N * 4);
static_assert(OFF_WB + (size_t)N_WB * 2 <= OFF_FC1, "wb must fit inside pool1 region");
static_assert((size_t)N_C2 * 2 <= AL((size_t)N_C1 * 2), "conv2 out must fit in conv1 region");

// ---------------------------------------------------------------------------
// conv1: 3->6, 5x5, stride 2, pad 1, relu. in fp32 (256,3,250,250) out bf16.
__global__ void k_conv1_relu(const float* __restrict__ x, const float* __restrict__ w,
                             const float* __restrict__ bias, bf16_t* __restrict__ out) {
    int idx = blockIdx.x * blockDim.x + threadIdx.x;
    if (idx >= N_C1) return;
    int ox = idx % C1_W; int t = idx / C1_W;
    int oy = t % C1_H;   t /= C1_H;
    int oc = t % C1_OC;  int b = t / C1_OC;
    float acc = bias[oc];
    const int iy0 = oy * 2 - 1, ix0 = ox * 2 - 1;
    const float* xb = x + (size_t)b * 3 * 250 * 250;
    const float* wo = w + oc * 75;
    for (int c = 0; c < 3; ++c) {
        const float* xc = xb + c * 62500;
        const float* wc = wo + c * 25;
        #pragma unroll
        for (int ky = 0; ky < 5; ++ky) {
            int iy = iy0 + ky;
            if ((unsigned)iy >= 250u) continue;
            const float* xr = xc + iy * 250;
            #pragma unroll
            for (int kx = 0; kx < 5; ++kx) {
                int ix = ix0 + kx;
                if ((unsigned)ix < 250u) acc = fmaf(xr[ix], wc[ky * 5 + kx], acc);
            }
        }
    }
    out[idx] = __float2bfloat16(fmaxf(acc, 0.0f));
}

// pool1: 2x2 stride-1 max, (124,124)->(123,123), bf16->bf16
__global__ void k_pool1(const bf16_t* __restrict__ in, bf16_t* __restrict__ out) {
    int idx = blockIdx.x * blockDim.x + threadIdx.x;
    if (idx >= N_P1) return;
    int xo = idx % P1_W; int t = idx / P1_W;
    int yo = t % P1_H;   int p = t / P1_H;               // p = b*6+c
    const bf16_t* s = in + ((size_t)p * C1_H + yo) * C1_W + xo;
    float m = __bfloat162float(s[0]);
    m = fmaxf(m, __bfloat162float(s[1]));
    m = fmaxf(m, __bfloat162float(s[C1_W]));
    m = fmaxf(m, __bfloat162float(s[C1_W + 1]));
    out[idx] = __float2bfloat16(m);
}

// conv2: 6->15, 3x3, stride 2, pad 1, relu. bf16 in, bf16 out.
__global__ void k_conv2_relu(const bf16_t* __restrict__ in, const float* __restrict__ w,
                             const float* __restrict__ bias, bf16_t* __restrict__ out) {
    int idx = blockIdx.x * blockDim.x + threadIdx.x;
    if (idx >= N_C2) return;
    int ox = idx % C2_W; int t = idx / C2_W;
    int oy = t % C2_H;   t /= C2_H;
    int oc = t % C2_OC;  int b = t / C2_OC;
    float acc = bias[oc];
    const int iy0 = oy * 2 - 1, ix0 = ox * 2 - 1;
    const bf16_t* ib = in + (size_t)b * C1_OC * P1_H * P1_W;
    const float* wo = w + oc * (6 * 9);
    for (int c = 0; c < 6; ++c) {
        const bf16_t* ic = ib + (size_t)c * P1_H * P1_W;
        const float* wc = wo + c * 9;
        #pragma unroll
        for (int ky = 0; ky < 3; ++ky) {
            int iy = iy0 + ky;
            if ((unsigned)iy >= (unsigned)P1_H) continue;
            const bf16_t* ir = ic + iy * P1_W;
            #pragma unroll
            for (int kx = 0; kx < 3; ++kx) {
                int ix = ix0 + kx;
                if ((unsigned)ix < (unsigned)P1_W)
                    acc = fmaf(__bfloat162float(ir[ix]), wc[ky * 3 + kx], acc);
            }
        }
    }
    out[idx] = __float2bfloat16(fmaxf(acc, 0.0f));
}

// pool2 + flatten + zero-pad K to KP. out: bf16 (256, KP), row k = c*3721+y*61+x
__global__ void k_pool2_pad(const bf16_t* __restrict__ in, bf16_t* __restrict__ out) {
    int idx = blockIdx.x * blockDim.x + threadIdx.x;
    if (idx >= N_HB) return;
    int k = idx % KP, b = idx / KP;
    if (k >= K_FLAT) { out[idx] = __float2bfloat16(0.0f); return; }
    int c = k / (P2_H * P2_W);
    int r = k % (P2_H * P2_W);
    int y = r / P2_W, x = r % P2_W;
    const bf16_t* s = in + (((size_t)b * C2_OC + c) * C2_H + y) * C2_W + x;
    float m = __bfloat162float(s[0]);
    m = fmaxf(m, __bfloat162float(s[1]));
    m = fmaxf(m, __bfloat162float(s[C2_W]));
    m = fmaxf(m, __bfloat162float(s[C2_W + 1]));
    out[idx] = __float2bfloat16(m);
}

// fc1 weight fp32 (120,55815) -> bf16 (128,KP) zero-padded
__global__ void k_cvt_w(const float* __restrict__ w, bf16_t* __restrict__ wb) {
    int idx = blockIdx.x * blockDim.x + threadIdx.x;
    if (idx >= N_WB) return;
    int k = idx % KP, n = idx / KP;
    float v = (n < FC1_N && k < K_FLAT) ? w[(size_t)n * K_FLAT + k] : 0.0f;
    wb[idx] = __float2bfloat16(v);
}

// ---------------------------------------------------------------------------
// fc1: C[m,n] = relu(sum_k h[m,k]*w[n,k] + b[n]), via v_wmma_f32_16x16x32_bf16
// Block = KW(=8) waves on one 16x16 output tile; wave wv owns 64-wide K-chunks
// k0 = wv*64, wv*64 + KW*64, ... (interleaved for L2 locality). Each wave
// accumulates privately (2 WMMAs / iter), then partials are reduced through
// LDS in fixed wave order (deterministic), bias+relu, wave 0 stores.
// A frag (16x32 bf16): lane(lo,hi) elems e<8 -> K=hi*8+e, e>=8 -> K=16+hi*8+(e-8)
// B frag (32x16 bf16): lane(lo,hi) elems e<8 -> K=hi*8+e, e>=8 -> K=16+hi*8+(e-8), column N=lo
// D (16x16 f32): elem r -> row m0+hi*8+r, col n0+lo
__global__ void k_fc1_wmma(const bf16_t* __restrict__ hb, const bf16_t* __restrict__ wb,
                           const float* __restrict__ bias, float* __restrict__ out) {
    __shared__ float red[KW * 32 * 8];
    const int lane = threadIdx.x & 31;
    const int wv   = threadIdx.x >> 5;
    const int lo = lane & 15, hi = lane >> 4;
    const int m0 = blockIdx.x * 16;
    const int n0 = blockIdx.y * 16;
    const bf16_t* pa = hb + (size_t)(m0 + lo) * KP + hi * 8;
    const bf16_t* pb = wb + (size_t)(n0 + lo) * KP + hi * 8;
    v8f acc = {};
    for (int k0 = wv * 64; k0 < KP; k0 += KW * 64) {
        const bf16_t* a0 = pa + k0;
        const bf16_t* b0 = pb + k0;
        union { v16bf m; u32x4 q[2]; } fa, fb, fa2, fb2;
        fa.q[0]  = *(const u32x4*)(a0);
        fa.q[1]  = *(const u32x4*)(a0 + 16);
        fb.q[0]  = *(const u32x4*)(b0);
        fb.q[1]  = *(const u32x4*)(b0 + 16);
        fa2.q[0] = *(const u32x4*)(a0 + 32);
        fa2.q[1] = *(const u32x4*)(a0 + 48);
        fb2.q[0] = *(const u32x4*)(b0 + 32);
        fb2.q[1] = *(const u32x4*)(b0 + 48);
        __builtin_prefetch(a0 + KW * 64, 0, 1);   // next chunk of this wave
        __builtin_prefetch(b0 + KW * 64, 0, 1);
        acc = __builtin_amdgcn_wmma_f32_16x16x32_bf16(
            false, fa.m, false, fb.m, (short)0, acc, false, false);
        acc = __builtin_amdgcn_wmma_f32_16x16x32_bf16(
            false, fa2.m, false, fb2.m, (short)0, acc, false, false);
    }
    // spill partials, deterministic fixed-order reduction by wave 0
    float* myred = &red[(wv * 32 + lane) * 8];
    #pragma unroll
    for (int r = 0; r < 8; ++r) myred[r] = acc[r];
    __syncthreads();
    if (wv == 0) {
        const int n = n0 + lo;
        if (n < FC1_N) {
            const float bn = bias[n];
            #pragma unroll
            for (int r = 0; r < 8; ++r) {
                float s = 0.0f;
                #pragma unroll
                for (int w8 = 0; w8 < KW; ++w8)   // fixed order -> deterministic
                    s += red[(w8 * 32 + lane) * 8 + r];
                int m = m0 + hi * 8 + r;
                out[(size_t)m * FC1_N + n] = fmaxf(s + bn, 0.0f);
            }
        }
    }
}

// fc2: (256,120)x(84,120)^T + b, relu (tiny: 5 MFLOP)
__global__ void k_fc2(const float* __restrict__ h, const float* __restrict__ w,
                      const float* __restrict__ bias, float* __restrict__ out) {
    int idx = blockIdx.x * blockDim.x + threadIdx.x;
    if (idx >= BATCH * FC2_N) return;
    int n = idx % FC2_N, b = idx / FC2_N;
    const float* hr = h + (size_t)b * FC1_N;
    const float* wr = w + (size_t)n * FC1_N;
    float acc = bias[n];
    #pragma unroll 4
    for (int k = 0; k < FC1_N; ++k) acc = fmaf(hr[k], wr[k], acc);
    out[idx] = fmaxf(acc, 0.0f);
}

// fc3 + 4-qubit statevector head + softmax. One thread per sample.
// wire q <-> index bit (3-q); RX(angles) per-sample, RY(qweights) shared, CZ chain.
__global__ void k_head(const float* __restrict__ h2, const float* __restrict__ w3,
                       const float* __restrict__ b3, const float* __restrict__ qw,
                       float* __restrict__ out) {
    int b = blockIdx.x * blockDim.x + threadIdx.x;
    if (b >= BATCH) return;
    const float* hr = h2 + (size_t)b * FC2_N;
    float ang[4];
    #pragma unroll
    for (int q = 0; q < 4; ++q) {
        float a = b3[q];
        for (int j = 0; j < FC2_N; ++j) a = fmaf(hr[j], w3[q * FC2_N + j], a);
        ang[q] = a;
    }
    float sr[16], si[16];
    #pragma unroll
    for (int i = 0; i < 16; ++i) { sr[i] = 0.0f; si[i] = 0.0f; }
    sr[0] = 1.0f;
    // RX encoding: [[c, -i s],[-i s, c]]
    #pragma unroll
    for (int q = 0; q < 4; ++q) {
        float t = ang[q] * 0.5f;
        float c = cosf(t), s = sinf(t);
        int st = 1 << (3 - q);
        #pragma unroll
        for (int i = 0; i < 16; ++i) {
            if (i & st) continue;
            int j = i | st;
            float a0r = sr[i], a0i = si[i], a1r = sr[j], a1i = si[j];
            sr[i] = c * a0r + s * a1i;  si[i] = c * a0i - s * a1r;
            sr[j] = c * a1r + s * a0i;  si[j] = c * a1i - s * a0r;
        }
    }
    int widx = 0;
    for (int d = 0; d < 2; ++d) {
        // RY: [[c,-s],[s,c]] (real)
        #pragma unroll
        for (int q = 0; q < 4; ++q) {
            float t = qw[widx++] * 0.5f;
            float c = cosf(t), s = sinf(t);
            int st = 1 << (3 - q);
            #pragma unroll
            for (int i = 0; i < 16; ++i) {
                if (i & st) continue;
                int j = i | st;
                float a0r = sr[i], a0i = si[i], a1r = sr[j], a1i = si[j];
                sr[i] = c * a0r - s * a1r;  si[i] = c * a0i - s * a1i;
                sr[j] = s * a0r + c * a1r;  si[j] = s * a0i + c * a1i;
            }
        }
        // CZ chain on (q, q+1)
        #pragma unroll
        for (int q = 0; q < 3; ++q) {
            int m = (1 << (3 - q)) | (1 << (2 - q));
            #pragma unroll
            for (int i = 0; i < 16; ++i)
                if ((i & m) == m) { sr[i] = -sr[i]; si[i] = -si[i]; }
        }
    }
    float z0 = 0.0f, z1 = 0.0f;
    #pragma unroll
    for (int i = 0; i < 16; ++i) {
        float p = sr[i] * sr[i] + si[i] * si[i];
        z0 += (i & 8) ? -p : p;
        z1 += (i & 4) ? -p : p;
    }
    float mx = fmaxf(z0, z1);
    float e0 = expf(z0 - mx), e1 = expf(z1 - mx);
    float inv = 1.0f / (e0 + e1);
    out[b * 2 + 0] = e0 * inv;
    out[b * 2 + 1] = e1 * inv;
}

// ---------------------------------------------------------------------------
extern "C" void kernel_launch(void* const* d_in, const int* in_sizes, int n_in,
                              void* d_out, int out_size, void* d_ws, size_t ws_size,
                              hipStream_t stream) {
    const float* x    = (const float*)d_in[0];
    const float* w1   = (const float*)d_in[1];
    const float* b1   = (const float*)d_in[2];
    const float* w2   = (const float*)d_in[3];
    const float* b2   = (const float*)d_in[4];
    const float* fc1w = (const float*)d_in[5];
    const float* fc1b = (const float*)d_in[6];
    const float* fc2w = (const float*)d_in[7];
    const float* fc2b = (const float*)d_in[8];
    const float* fc3w = (const float*)d_in[9];
    const float* fc3b = (const float*)d_in[10];
    const float* qw   = (const float*)d_in[11];
    char* ws = (char*)d_ws;

    bf16_t* c1  = (bf16_t*)(ws + OFF_C1);   // conv1 out
    bf16_t* p1  = (bf16_t*)(ws + OFF_P1);   // pool1 out
    bf16_t* c2  = (bf16_t*)(ws + OFF_C1);   // conv2 out (reuses conv1 region)
    bf16_t* hb  = (bf16_t*)(ws + OFF_P1);   // fc1 activations (reuses pool1 region)
    bf16_t* wbf = (bf16_t*)(ws + OFF_WB);   // fc1 weights bf16
    float*  f1  = (float*)(ws + OFF_FC1);   // fc1 out
    float*  f2  = (float*)(ws + OFF_FC2);   // fc2 out
    float*  outp = (float*)d_out;

    const int T = 256;
    k_conv1_relu<<<(N_C1 + T - 1) / T, T, 0, stream>>>(x, w1, b1, c1);
    k_pool1     <<<(N_P1 + T - 1) / T, T, 0, stream>>>(c1, p1);
    k_conv2_relu<<<(N_C2 + T - 1) / T, T, 0, stream>>>(p1, w2, b2, c2);
    // weight conversion only after pool1 region is dead (wbf overlaps it)
    k_cvt_w     <<<(N_WB + T - 1) / T, T, 0, stream>>>(fc1w, wbf);
    k_pool2_pad <<<(N_HB + T - 1) / T, T, 0, stream>>>(c2, hb);
    k_fc1_wmma  <<<dim3(BATCH / 16, 8), KW * 32, 0, stream>>>(hb, wbf, fc1b, f1);
    k_fc2       <<<(BATCH * FC2_N + T - 1) / T, T, 0, stream>>>(f1, fc2w, fc2b, f2);
    k_head      <<<1, BATCH, 0, stream>>>(f2, fc3w, fc3b, qw, outp);
}
